// a3c_lstm_ga_attention_single_50929722196546
// MI455X (gfx1250) — matrix operand order repaired and checked by reading, not compiled, not run on hardware
//
#include <hip/hip_runtime.h>
#include <hip/hip_bf16.h>
#include <math.h>

typedef __attribute__((ext_vector_type(16))) _Float16 v16h;
typedef __attribute__((ext_vector_type(8)))  float    v8f;

#define DIVUP(a,b) (((a)+(b)-1)/(b))

// ---------------------------------------------------------------------------
// WMMA f16 register layouts used here (C/D as in ISA 7.12.2, wave32):
//  A (16x32, MxK): lane L -> M=L&15 ; half j -> K = j + 16*(L>>4) (this code's order, not ISA table)
//  B (32x16, KxN): lane L -> N=L&15 ; half j -> K = j + 16*(L>>4) (same order as A, not ISA table)
//  C/D (16x16 f32): lane L, vgpr r -> M = r + 8*(L>>4), N = L&15
// ---------------------------------------------------------------------------

__device__ __forceinline__ int a_kidx(int lane, int j) {
    return j + 16 * (lane >> 4);
}

// Swizzle f32 row-major [Mrows x ldk] -> f16 A-tiles [(mt*Kt+kt)*512 + lane*16 + j]
__global__ void swizzleA_k(const float* __restrict__ src, _Float16* __restrict__ dst,
                           int Mt, int Kt, int ldk, int Mrows) {
    int tid = blockIdx.x * blockDim.x + threadIdx.x;
    int total = Mt * Kt * 512;
    if (tid >= total) return;
    int j = tid & 15, lane = (tid >> 4) & 31, tile = tid >> 9;
    int mt = tile / Kt, kt = tile % Kt;
    int m = mt * 16 + (lane & 15);
    int K = kt * 32 + a_kidx(lane, j);
    float v = (m < Mrows) ? src[(size_t)m * ldk + K] : 0.f;
    dst[tid] = (_Float16)v;
}

// Weight [N x ldk] row-major, used as B = Wt (KxN): B[k][n] = W[n][k]
__global__ void swizzleB_k(const float* __restrict__ src, _Float16* __restrict__ dst,
                           int Nt, int Kt, int ldk) {
    int tid = blockIdx.x * blockDim.x + threadIdx.x;
    int total = Nt * Kt * 512;
    if (tid >= total) return;
    int j = tid & 15, lane = (tid >> 4) & 31, tile = tid >> 9;
    int nt = tile / Kt, kt = tile % Kt;
    int n = nt * 16 + (lane & 15);
    int K = kt * 32 + j + 16 * (lane >> 4);
    dst[tid] = (_Float16)src[(size_t)n * ldk + K];
}

// Token embedding gather fused with A-swizzle: M=192 tokens, K=32 emb dim
__global__ void emb_swz_k(const float* __restrict__ emb_w,
                          const int* __restrict__ prev, const int* __restrict__ curr,
                          const int* __restrict__ next, _Float16* __restrict__ dst) {
    int tid = blockIdx.x * blockDim.x + threadIdx.x;
    if (tid >= 12 * 512) return;
    int j = tid & 15, lane = (tid >> 4) & 31, mt = tid >> 9;   // Kt == 1
    int m = mt * 16 + (lane & 15);
    int K = a_kidx(lane, j);
    int tok = (m < 64) ? prev[m] : (m < 128) ? curr[m - 64] : next[m - 128];
    dst[tid] = (_Float16)emb_w[(size_t)tok * 32 + K];
}

// im2col fused with A-swizzle.  src: f32 NCHW [C][H][W]; K-dim = c*ksz*ksz+kh*ksz+kw
__global__ void im2col_k(const float* __restrict__ src, _Float16* __restrict__ dst,
                         int C, int H, int W, int ksz, int stride,
                         int OH, int OW, int Mt, int Kt) {
    int tid = blockIdx.x * blockDim.x + threadIdx.x;
    int total = Mt * Kt * 512;
    if (tid >= total) return;
    int j = tid & 15, lane = (tid >> 4) & 31, tile = tid >> 9;
    int mt = tile / Kt, kt = tile % Kt;
    int m = mt * 16 + (lane & 15);
    int K = kt * 32 + a_kidx(lane, j);
    float v = 0.f;
    if (m < OH * OW) {
        int kk = ksz * ksz;
        int c = K / kk, rem = K - c * kk;
        int kh = rem / ksz, kw = rem - kh * ksz;
        int oh = m / OW, ow = m - oh * OW;
        v = src[(size_t)c * H * W + (size_t)(oh * stride + kh) * W + (ow * stride + kw)];
    }
    dst[tid] = (_Float16)v;
}

// ---------------------------------------------------------------------------
// Generic WMMA GEMM on pre-swizzled f16 tiles.  One wave per 16x16 output tile.
// out = act(A*B + bias[n]);  transpose=1 -> out[n*ldout+m] (NCHW conv output)
// ---------------------------------------------------------------------------
__global__ __launch_bounds__(32) void gemm_wmma_k(
    const _Float16* __restrict__ A, const _Float16* __restrict__ B,
    const float* __restrict__ bias, float* __restrict__ out,
    int Mt, int Nt, int Kt, int Mrows, int act, int transpose, int ldout) {
    int bx = blockIdx.x;
    int mt = bx / Nt, nt = bx % Nt;
    int lane = threadIdx.x;
    const _Float16* ap = A + (size_t)(mt * Kt) * 512 + lane * 16;
    const _Float16* bp = B + (size_t)(nt * Kt) * 512 + lane * 16;
    v8f c = {};
    for (int kt = 0; kt < Kt; ++kt) {
        v16h a = *(const v16h*)ap;
        v16h b = *(const v16h*)bp;
        __builtin_prefetch(ap + 512, 0, 0);   // speculative global_prefetch for next A tile
        ap += 512; bp += 512;
        c = __builtin_amdgcn_wmma_f32_16x16x32_f16(false, a, false, b, (short)0, c,
                                                   false, false);
    }
    int n = nt * 16 + (lane & 15);
    int mbase = mt * 16 + ((lane >> 4) << 3);
    float bv = bias ? bias[n] : 0.f;
#pragma unroll
    for (int r = 0; r < 8; ++r) {
        int m = mbase + r;
        float v = c[r] + bv;
        if (act) v = v > 0.f ? v : 0.f;
        if (m < Mrows) {
            if (transpose) out[(size_t)n * ldout + m] = v;
            else           out[(size_t)m * ldout + n] = v;
        }
    }
}

// ---------------------------------------------------------------------------
// GRU scan: 2 workgroups (fwd / "bwd"-same-order), 8 waves each.
// Per step: gh = whh @ h as WMMA matvec (h in B column 0), then gate math.
// kt-outer loop: B operand built ONCE per K-slice, streamed against 6 A tiles
// held in 6 accumulators (48 VGPRs) -> 6x fewer DS ops in the serial chain.
// ---------------------------------------------------------------------------
__global__ __launch_bounds__(256) void gru_scan_k(
    const float* __restrict__ giF, const float* __restrict__ giB,
    const _Float16* __restrict__ whhF, const _Float16* __restrict__ whhB,
    const float* __restrict__ bhhF, const float* __restrict__ bhhB,
    float* __restrict__ hsF, float* __restrict__ hsB) {
    const float*    gi  = blockIdx.x ? giB  : giF;
    const _Float16* whh = blockIdx.x ? whhB : whhF;
    const float*    bhh = blockIdx.x ? bhhB : bhhF;
    float*          hs  = blockIdx.x ? hsB  : hsF;

    __shared__ _Float16 hsh[256];   // h in f16 for WMMA B operand
    __shared__ float    gh[768];    // whh @ h

    int i = threadIdx.x;
    int lane = i & 31, wave = i >> 5;
    int nn = lane & 15, g = lane >> 4;
    float h = 0.f;
    hsh[i] = (_Float16)0.f;
    __syncthreads();

    // wave owns M-tiles [wave*6, wave*6+6); A tile (mt,kt) at (mt*8+kt)*512
    const _Float16* abase = whh + (size_t)(wave * 6 * 8) * 512 + lane * 16;

    for (int t = 0; t < 192; ++t) {
        v8f acc[6];
#pragma unroll
        for (int mi = 0; mi < 6; ++mi) acc[mi] = (v8f){};
#pragma unroll
        for (int kt = 0; kt < 8; ++kt) {
            v16h b;
#pragma unroll
            for (int j = 0; j < 16; ++j) {
                _Float16 hv = hsh[kt * 32 + j + 16 * g];   // broadcast DS read
                b[j] = (nn == 0) ? hv : (_Float16)0.f;     // h lives in column 0
            }
#pragma unroll
            for (int mi = 0; mi < 6; ++mi) {
                v16h a = *(const v16h*)(abase + (size_t)(mi * 8 + kt) * 512);
                acc[mi] = __builtin_amdgcn_wmma_f32_16x16x32_f16(
                    false, a, false, b, (short)0, acc[mi], false, false);
            }
        }
        if (nn == 0) {                                      // column 0 holds result
#pragma unroll
            for (int mi = 0; mi < 6; ++mi) {
                int mb = (wave * 6 + mi) * 16 + 8 * g;
#pragma unroll
                for (int r = 0; r < 8; ++r) gh[mb + r] = acc[mi][r];
            }
        }
        __syncthreads();
        // ---- gates (thread i owns h element i) ----
        const float* git = gi + (size_t)t * 768;
        float rr = 1.f / (1.f + expf(-(git[i]       + gh[i]       + bhh[i])));
        float zz = 1.f / (1.f + expf(-(git[256 + i] + gh[256 + i] + bhh[256 + i])));
        float nv = tanhf(git[512 + i] + rr * (gh[512 + i] + bhh[512 + i]));
        h = (1.f - zz) * nv + zz * h;
        hsh[i] = (_Float16)h;
        hs[(size_t)t * 256 + i] = h;
        __syncthreads();
    }
}

// ---------------------------------------------------------------------------
// Fused head: image emb, attention, bilinear softmax pooling, gating, LSTM cell,
// critic/actor.  Single workgroup, 256 threads, LDS-staged (all tiny matvecs).
// ---------------------------------------------------------------------------
__global__ __launch_bounds__(256) void head_k(
    const float* __restrict__ hsF, const float* __restrict__ hsB,
    const float* __restrict__ c3,                       // [2304] flat (c,h,w)
    const float* __restrict__ hx, const float* __restrict__ cx,
    const float* __restrict__ img_w, const float* __restrict__ img_b,
    const float* __restrict__ key_w, const float* __restrict__ key_b,
    const float* __restrict__ bilA,  const float* __restrict__ bilb,
    const float* __restrict__ red_w, const float* __restrict__ red_b,
    const float* __restrict__ gatw,  const float* __restrict__ gatb,
    const float* __restrict__ lin_w, const float* __restrict__ lin_b,
    const float* __restrict__ lwih,  const float* __restrict__ lwhh,
    const float* __restrict__ lbih,  const float* __restrict__ lbhh,
    const float* __restrict__ cri_w, const float* __restrict__ cri_b,
    const float* __restrict__ act_w, const float* __restrict__ act_b,
    const float* __restrict__ time_w, const float* __restrict__ temp_w,
    const int* __restrict__ tx, float* __restrict__ out) {
    __shared__ float cir[512], img[256], key[256], u[544], sc[128],
                     isum[544], gating[128], gkey[64], zz[256],
                     h2[256], c2[256], feat[288], red[1];
    int i = threadIdx.x;
    cir[i]       = hsF[127 * 256 + i];                 // hs_f[np+nc-1]
    cir[256 + i] = hsB[127 * 256 + i];                 // hs_b reversed [np] -> raw [127]
    { // image embedding
        float a = img_b[i];
        const float* wr = img_w + (size_t)i * 2304;
        for (int k = 0; k < 2304; ++k) a += wr[k] * c3[k];
        img[i] = a;
    }
    __syncthreads();
    { // attention key
        float a = key_b[i];
        const float* wr = key_w + (size_t)i * 768;
        for (int k = 0; k < 512; ++k) a += wr[k] * cir[k];
        for (int k = 0; k < 256; ++k) a += wr[512 + k] * img[k];
        key[i] = a;
    }
    __syncthreads();
    for (int v = i; v < 544; v += 256) {               // u = key @ bilinear_A
        float a = 0.f;
        for (int k = 0; k < 256; ++k) a += key[k] * bilA[(size_t)k * 544 + v];
        u[v] = a;
    }
    __syncthreads();
    if (i < 128) {                                     // scores[m] = u . rep[m]
        int tok = (i < 64) ? i : i + 64;
        int ts  = (i < 64) ? 0 : 1;
        const float* hf = hsF + (size_t)tok * 256;
        const float* hb = hsB + (size_t)(191 - tok) * 256;
        float a = bilb[0];
        for (int v = 0; v < 256; ++v) a += u[v] * hf[v];
        for (int v = 0; v < 256; ++v) a += u[256 + v] * hb[v];
        for (int v = 0; v < 32;  ++v) a += u[512 + v] * temp_w[ts * 32 + v];
        sc[i] = a;
    }
    __syncthreads();
    if (i == 0) {                                      // softmax over 128 (tiny)
        float mx = sc[0];
        for (int m = 1; m < 128; ++m) mx = sc[m] > mx ? sc[m] : mx;
        float s = 0.f;
        for (int m = 0; m < 128; ++m) { float e = expf(sc[m] - mx); sc[m] = e; s += e; }
        red[0] = 1.f / s;
    }
    __syncthreads();
    float invs = red[0];
    for (int v = i; v < 544; v += 256) {               // weighted instr sum
        float a = 0.f;
        for (int m = 0; m < 128; ++m) {
            int tok = (m < 64) ? m : m + 64;
            float rep;
            if (v < 256)      rep = hsF[(size_t)tok * 256 + v];
            else if (v < 512) rep = hsB[(size_t)(191 - tok) * 256 + (v - 256)];
            else              rep = temp_w[((m < 64) ? 0 : 1) * 32 + (v - 512)];
            a += sc[m] * invs * rep;
        }
        isum[v] = a;
    }
    __syncthreads();
    if (i < 128) {                                     // gating = reduce(instr_sum)
        float a = red_b[i];
        const float* wr = red_w + (size_t)i * 544;
        for (int v = 0; v < 544; ++v) a += wr[v] * isum[v];
        gating[i] = a;
    }
    __syncthreads();
    if (i < 64) {                                      // gate key (sigmoid)
        float a = gatb[i];
        const float* wr = gatw + (size_t)i * 640;
        for (int k = 0; k < 512; ++k) a += wr[k] * cir[k];
        for (int k = 0; k < 128; ++k) a += wr[512 + k] * gating[k];
        gkey[i] = 1.f / (1.f + expf(-a));
    }
    __syncthreads();
    { // z = relu(lin(x_img * gate))
        float a = lin_b[i];
        const float* wr = lin_w + (size_t)i * 2304;
        for (int k = 0; k < 2304; ++k) a += wr[k] * c3[k] * gkey[k / 36];
        zz[i] = a > 0.f ? a : 0.f;
    }
    __syncthreads();
    { // LSTMCell (torch gate order i,f,g,o)
        float g4[4];
        for (int q = 0; q < 4; ++q) {
            int r = q * 256 + i;
            float a = lbih[r] + lbhh[r];
            const float* wi = lwih + (size_t)r * 256;
            const float* wh = lwhh + (size_t)r * 256;
            for (int k = 0; k < 256; ++k) a += wi[k] * zz[k] + wh[k] * hx[k];
            g4[q] = a;
        }
        float ig = 1.f / (1.f + expf(-g4[0]));
        float fg = 1.f / (1.f + expf(-g4[1]));
        float gg = tanhf(g4[2]);
        float og = 1.f / (1.f + expf(-g4[3]));
        float cn = fg * cx[i] + ig * gg;
        float hn = og * tanhf(cn);
        c2[i] = cn; h2[i] = hn; feat[i] = hn;
    }
    if (i < 32) feat[256 + i] = time_w[(size_t)tx[0] * 32 + i];
    __syncthreads();
    if (i < 5) {                                       // critic[1] + actor[4]
        if (i == 0) {
            float a = cri_b[0];
            for (int k = 0; k < 288; ++k) a += cri_w[k] * feat[k];
            out[0] = a;
        } else {
            int r = i - 1;
            float a = act_b[r];
            for (int k = 0; k < 288; ++k) a += act_w[(size_t)r * 288 + k] * feat[k];
            out[1 + r] = a;
        }
    }
    out[5 + i]   = h2[i];
    out[261 + i] = c2[i];
}

// ---------------------------------------------------------------------------
extern "C" void kernel_launch(void* const* d_in, const int* in_sizes, int n_in,
                              void* d_out, int out_size, void* d_ws, size_t ws_size,
                              hipStream_t stream) {
    (void)in_sizes; (void)n_in; (void)out_size; (void)ws_size;
    const float* x        = (const float*)d_in[0];
    const float* hx       = (const float*)d_in[1];
    const float* cx       = (const float*)d_in[2];
    const float* conv1_w  = (const float*)d_in[3];
    const float* conv1_b  = (const float*)d_in[4];
    const float* conv2_w  = (const float*)d_in[5];
    const float* conv2_b  = (const float*)d_in[6];
    const float* conv3_w  = (const float*)d_in[7];
    const float* conv3_b  = (const float*)d_in[8];
    const float* emb_w    = (const float*)d_in[9];
    const float* gf_wih   = (const float*)d_in[10];
    const float* gf_whh   = (const float*)d_in[11];
    const float* gf_bih   = (const float*)d_in[12];
    const float* gf_bhh   = (const float*)d_in[13];
    const float* gb_wih   = (const float*)d_in[14];
    const float* gb_whh   = (const float*)d_in[15];
    const float* gb_bih   = (const float*)d_in[16];
    const float* gb_bhh   = (const float*)d_in[17];
    const float* time_w   = (const float*)d_in[18];
    const float* temp_w   = (const float*)d_in[19];
    const float* img_w    = (const float*)d_in[20];
    const float* img_b    = (const float*)d_in[21];
    const float* key_w    = (const float*)d_in[22];
    const float* key_b    = (const float*)d_in[23];
    const float* bilA     = (const float*)d_in[24];
    const float* bilb     = (const float*)d_in[25];
    const float* red_w    = (const float*)d_in[26];
    const float* red_b    = (const float*)d_in[27];
    const float* gat_w    = (const float*)d_in[28];
    const float* gat_b    = (const float*)d_in[29];
    const float* lin_w    = (const float*)d_in[30];
    const float* lin_b    = (const float*)d_in[31];
    const float* lwih     = (const float*)d_in[32];
    const float* lwhh     = (const float*)d_in[33];
    const float* lbih     = (const float*)d_in[34];
    const float* lbhh     = (const float*)d_in[35];
    const float* cri_w    = (const float*)d_in[36];
    const float* cri_b    = (const float*)d_in[37];
    const float* act_w    = (const float*)d_in[38];
    const float* act_b    = (const float*)d_in[39];
    const int*   prev     = (const int*)d_in[40];
    const int*   curr     = (const int*)d_in[41];
    const int*   next     = (const int*)d_in[42];
    const int*   tx       = (const int*)d_in[43];
    float* out = (float*)d_out;

    char* ws = (char*)d_ws;
    _Float16* whhF_s = (_Float16*)(ws + 0);          // 48x8 tiles  (393216 B)
    _Float16* whhB_s = (_Float16*)(ws + 393216);
    _Float16* wihF_s = (_Float16*)(ws + 786432);     // 48x1 tiles  (49152 B)
    _Float16* wihB_s = (_Float16*)(ws + 835584);
    _Float16* w1_s   = (_Float16*)(ws + 884736);     // 8x6 tiles
    _Float16* w2_s   = (_Float16*)(ws + 933888);     // 4x64 tiles
    _Float16* w3_s   = (_Float16*)(ws + 1196032);    // 4x32 tiles
    _Float16* emb_s  = (_Float16*)(ws + 1327104);    // 12x1 tiles
    _Float16* im2c   = (_Float16*)(ws + 1339392);    // reused, max 13x64 tiles (832 KB)
    float*    c1o    = (float*)   (ws + 2191360);    // [128][900]
    float*    c2o    = (float*)   (ws + 2652160);    // [64][196]
    float*    c3o    = (float*)   (ws + 2702336);    // [64][36] = x_img flat
    float*    giF    = (float*)   (ws + 2711552);    // [192][768]
    float*    giB    = (float*)   (ws + 3301376);
    float*    hsF    = (float*)   (ws + 3891200);    // [192][256]
    float*    hsB    = (float*)   (ws + 4087808);

    // ---- weight prep (f32 -> f16, pre-swizzled into WMMA register layout) ----
    swizzleA_k<<<DIVUP(48*8*512,256),256,0,stream>>>(gf_whh, whhF_s, 48, 8, 256, 768);
    swizzleA_k<<<DIVUP(48*8*512,256),256,0,stream>>>(gb_whh, whhB_s, 48, 8, 256, 768);
    swizzleB_k<<<DIVUP(48*1*512,256),256,0,stream>>>(gf_wih, wihF_s, 48, 1, 32);
    swizzleB_k<<<DIVUP(48*1*512,256),256,0,stream>>>(gb_wih, wihB_s, 48, 1, 32);
    swizzleB_k<<<DIVUP(8*6*512,256),256,0,stream>>>(conv1_w, w1_s, 8, 6, 192);
    swizzleB_k<<<DIVUP(4*64*512,256),256,0,stream>>>(conv2_w, w2_s, 4, 64, 2048);
    swizzleB_k<<<DIVUP(4*32*512,256),256,0,stream>>>(conv3_w, w3_s, 4, 32, 1024);
    emb_swz_k <<<DIVUP(12*512,256),256,0,stream>>>(emb_w, prev, curr, next, emb_s);

    // ---- CNN encoder as implicit GEMM (im2col fused with A-swizzle) ----
    im2col_k<<<DIVUP(57*6*512,256),256,0,stream>>>(x,   im2c, 3,   124, 124, 8, 4, 30, 30, 57, 6);
    gemm_wmma_k<<<57*8, 32, 0, stream>>>(im2c, w1_s, conv1_b, c1o, 57, 8, 6,  900, 1, 1, 900);
    im2col_k<<<DIVUP(13*64*512,256),256,0,stream>>>(c1o, im2c, 128, 30,  30,  4, 2, 14, 14, 13, 64);
    gemm_wmma_k<<<13*4, 32, 0, stream>>>(im2c, w2_s, conv2_b, c2o, 13, 4, 64, 196, 1, 1, 196);
    im2col_k<<<DIVUP(3*32*512,256),256,0,stream>>>(c2o, im2c, 64,  14,  14,  4, 2, 6,  6,  3,  32);
    gemm_wmma_k<<<3*4, 32, 0, stream>>>(im2c, w3_s, conv3_b, c3o, 3, 4, 32, 36, 1, 1, 36);

    // ---- GRU input projections: gi = emb @ wih^T + bih  (M=192, N=768, K=32) ----
    gemm_wmma_k<<<12*48, 32, 0, stream>>>(emb_s, wihF_s, gf_bih, giF, 12, 48, 1, 192, 0, 0, 768);
    gemm_wmma_k<<<12*48, 32, 0, stream>>>(emb_s, wihB_s, gb_bih, giB, 12, 48, 1, 192, 0, 0, 768);

    // ---- recurrent scans (both directions concurrently, WMMA matvec per step) ----
    gru_scan_k<<<2, 256, 0, stream>>>(giF, giB, whhF_s, whhB_s, gf_bhh, gb_bhh, hsF, hsB);

    // ---- fused attention / gating / LSTM / heads ----
    head_k<<<1, 256, 0, stream>>>(hsF, hsB, c3o, hx, cx,
                                  img_w, img_b, key_w, key_b, bilA, bilb,
                                  red_w, red_b, gat_w, gat_b, lin_w, lin_b,
                                  lwih, lwhh, lbih, lbhh,
                                  cri_w, cri_b, act_w, act_b,
                                  time_w, temp_w, tx, out);
}
